// MHA_42391327211690
// MI455X (gfx1250) — compile-verified, matrix operand order checked by reading, not run on hardware
//
#include <hip/hip_runtime.h>
#include <hip/hip_bf16.h>

// Problem constants (match the reference)
#define BATCH 1
#define SEQ   2048
#define EMB   2048
#define NH    32
#define NKV   8
#define HDIM  64
#define OPROJ (HDIM * (NH + 2 * NKV))   // 3072
#define HD    (NH * HDIM)               // 2048

typedef __attribute__((ext_vector_type(16))) __bf16 v16bf;
typedef __attribute__((ext_vector_type(8)))  float  v8f;

union FragU { v16bf v; uint4 q[2]; };

// ---------------------------------------------------------------------------
// Load a 16x32 bf16 WMMA fragment where each "row" (A: M-row, B: N-column) is
// contiguous in memory with leading dimension ld.
// ISA layout (16-bit A/B 16x32): lanes 0-15 hold K=0..7 (v[0..7]) and K=16..23
// (v[8..15]); lanes 16-31 hold K=8..15 and K=24..31.
// ---------------------------------------------------------------------------
__device__ __forceinline__ v16bf load_frag_rowK(const __bf16* base, int row0,
                                                int ld, int k0) {
  const int lane = threadIdx.x & 31;
  const int half = lane >> 4;
  const int ln   = lane & 15;
  const __bf16* p = base + (size_t)(row0 + ln) * (size_t)ld + k0 + half * 8;
  FragU f;
  f.q[0] = *(const uint4*)(p);        // K = half*8 .. half*8+7
  f.q[1] = *(const uint4*)(p + 16);   // K = 16+half*8 .. +7
  return f.v;
}

// B fragment with strided K (V matrix: B[k=t,n=d], stored V[t*ld + d]).
__device__ __forceinline__ v16bf load_frag_colgather(const __bf16* base, int k0,
                                                     int ld, int col0) {
  const int lane = threadIdx.x & 31;
  const int half = lane >> 4;
  const int ln   = lane & 15;
  v16bf v;
#pragma unroll
  for (int e = 0; e < 16; ++e) {
    const int kk = (e < 8) ? (half * 8 + e) : (16 + half * 8 + (e - 8));
    v[e] = base[(size_t)(k0 + kk) * (size_t)ld + col0 + ln];
  }
  return v;
}

__device__ __forceinline__ v8f wmma_bf16(v16bf a, v16bf b, v8f c) {
  return __builtin_amdgcn_wmma_f32_16x16x32_bf16(false, a, false, b,
                                                 (short)0, c, false, false);
}

// ---------------------------------------------------------------------------
// Tensor Data Mover: async 2-D tile load global->LDS (CDNA5 TDM, TENSORcnt).
// Builds a D# per CDNA5 ISA: count=1, type=2 ("image"),
// data_size=2 bytes, tile = tile_cols x tile_rows, row stride in elements.
// This toolchain exposes the 6-arg builtin:
//   (uint32x4 g0, int32x8 g1, int32x4 g2, int32x4 g3, int32x8 pad, i32 cpol)
// ---------------------------------------------------------------------------
#if __has_builtin(__builtin_amdgcn_tensor_load_to_lds) && \
    __has_builtin(__builtin_amdgcn_s_wait_tensorcnt)
#define HAVE_TDM 1
#else
#define HAVE_TDM 0
#endif

__device__ __forceinline__ void tdm_load_tile_2d(unsigned lds_off,
                                                 const void* gptr,
                                                 unsigned tile_rows,
                                                 unsigned tile_cols,
                                                 unsigned row_stride) {
#if HAVE_TDM
  typedef unsigned int v4u __attribute__((ext_vector_type(4)));
  typedef int v8i __attribute__((ext_vector_type(8)));
  typedef int v4i __attribute__((ext_vector_type(4)));
  const unsigned long long ga = (unsigned long long)gptr;
  v4u g0;
  g0[0] = 1u;                                     // count=1, user mode
  g0[1] = lds_off;                                // lds_addr [63:32]
  g0[2] = (unsigned)(ga & 0xFFFFFFFFu);           // global_addr lo
  g0[3] = (unsigned)((ga >> 32) & 0x1FFFFFFu)     // global_addr hi (57-bit)
          | (2u << 30);                           // type=2 (image)
  v8i g1;
  g1[0] = (int)(1u << 16);                        // data_size=1 -> 2 bytes
  g1[1] = (int)((tile_cols & 0xFFFFu) << 16);     // tensor_dim0 lo16 @ [63:48]
  g1[2] = (int)((tile_rows & 0xFFFFu) << 16);     // tensor_dim1 lo16 @ [95:80]
  g1[3] = (int)((tile_cols & 0xFFFFu) << 16);     // tile_dim0 @ [127:112]
  g1[4] = (int)(tile_rows & 0xFFFFu);             // tile_dim1 @ [143:128]
  g1[5] = (int)row_stride;                        // tensor_dim0_stride lo32
  g1[6] = 0;
  g1[7] = 0;
  const v4i z4 = {0, 0, 0, 0};
  const v8i z8 = {0, 0, 0, 0, 0, 0, 0, 0};
  __builtin_amdgcn_tensor_load_to_lds(g0, g1, z4, z4, z8, 0);
#else
  (void)lds_off; (void)gptr; (void)tile_rows; (void)tile_cols; (void)row_stride;
#endif
}

__device__ __forceinline__ void tdm_wait0() {
#if HAVE_TDM
  __builtin_amdgcn_s_wait_tensorcnt((short)0);
#endif
}

// Low 32 bits of a generic pointer to LDS == LDS byte offset (ISA aperture rule)
__device__ __forceinline__ unsigned lds_offset_of(const void* p) {
  return (unsigned)(unsigned long long)(uintptr_t)p;
}

// ---------------------------------------------------------------------------
// fp32 -> bf16 cast (grid-stride)
// ---------------------------------------------------------------------------
__global__ void mha_cvt_bf16(const float* __restrict__ in,
                             __bf16* __restrict__ out, int n) {
  for (int i = blockIdx.x * blockDim.x + threadIdx.x; i < n;
       i += gridDim.x * blockDim.x) {
    out[i] = (__bf16)in[i];
  }
}

// ---------------------------------------------------------------------------
// C[M,N] = A[M,K] * Bw[N,K]^T + bias[N]  (bf16 in, fp32 out)
// Block: 128 threads (4 waves); block tile 64x128; each wave 32x64 via a
// 2x4 grid of 16x16 WMMA accumulators, K-step 32 (8 WMMAs / 6 frag loads).
// ---------------------------------------------------------------------------
__global__ void mha_gemm_bf16(const __bf16* __restrict__ A,
                              const __bf16* __restrict__ Bw,
                              const float* __restrict__ bias,
                              float* __restrict__ C, int M, int N, int K) {
  const int wave = threadIdx.x >> 5;
  const int half = (threadIdx.x >> 4) & 1;
  const int ln   = threadIdx.x & 15;

  const int rbase = blockIdx.y * 64 + (wave >> 1) * 32;
  const int cbase = blockIdx.x * 128 + (wave & 1) * 64;

  v8f acc[2][4];
#pragma unroll
  for (int i = 0; i < 2; ++i)
#pragma unroll
    for (int j = 0; j < 4; ++j) acc[i][j] = (v8f){};

  for (int k0 = 0; k0 < K; k0 += 32) {
    if (k0 + 32 < K) {  // global_prefetch_b8 next K tiles
      __builtin_prefetch(A + (size_t)(rbase + ln) * K + k0 + 32, 0, 3);
      __builtin_prefetch(Bw + (size_t)(cbase + ln) * K + k0 + 32, 0, 3);
      __builtin_prefetch(Bw + (size_t)(cbase + 32 + ln) * K + k0 + 32, 0, 3);
    }
    v16bf a0 = load_frag_rowK(A, rbase,      K, k0);
    v16bf a1 = load_frag_rowK(A, rbase + 16, K, k0);
    v16bf b[4];
#pragma unroll
    for (int j = 0; j < 4; ++j) b[j] = load_frag_rowK(Bw, cbase + j * 16, K, k0);
#pragma unroll
    for (int j = 0; j < 4; ++j) {
      acc[0][j] = wmma_bf16(a0, b[j], acc[0][j]);
      acc[1][j] = wmma_bf16(a1, b[j], acc[1][j]);
    }
  }

#pragma unroll
  for (int i = 0; i < 2; ++i)
#pragma unroll
    for (int j = 0; j < 4; ++j)
#pragma unroll
      for (int r = 0; r < 8; ++r) {
        const int row = rbase + i * 16 + r + 8 * half;
        const int col = cbase + j * 16 + ln;
        C[(size_t)row * N + col] = acc[i][j][r] + bias[col];
      }
}

// ---------------------------------------------------------------------------
// RoPE + QKV split: qkv fp32 [S, OPROJ] -> head-major bf16 Q[NH][S][D],
// K[NKV][S][D] (rotated), V[NKV][S][D].
// ---------------------------------------------------------------------------
__global__ void mha_rope_split(const float* __restrict__ qkv,
                               __bf16* __restrict__ Qb,
                               __bf16* __restrict__ Kb,
                               __bf16* __restrict__ Vb) {
  const int idx = blockIdx.x * blockDim.x + threadIdx.x;
  if (idx >= SEQ * OPROJ) return;
  const int t = idx / OPROJ;
  const int o = idx % OPROJ;
  const float val = qkv[idx];

  if (o < NH * HDIM) {                       // ---- Q, with RoPE
    const int h = o / HDIM, d = o % HDIM;
    const size_t rowb = (size_t)t * OPROJ + (size_t)h * HDIM;
    float outv;
    if (d < 32) {
      const float th = (float)t * __powf(10000.0f, -(float)d / 32.0f);
      outv = val * __cosf(th) - qkv[rowb + d + 32] * __sinf(th);
    } else {
      const int i = d - 32;
      const float th = (float)t * __powf(10000.0f, -(float)i / 32.0f);
      outv = qkv[rowb + i] * __sinf(th) + val * __cosf(th);
    }
    Qb[(size_t)h * SEQ * HDIM + (size_t)t * HDIM + d] = (__bf16)outv;
  } else if (o < (NH + NKV) * HDIM) {        // ---- K, with RoPE
    const int oo = o - NH * HDIM;
    const int hk = oo / HDIM, d = oo % HDIM;
    const size_t rowb = (size_t)t * OPROJ + (size_t)(NH + hk) * HDIM;
    float outv;
    if (d < 32) {
      const float th = (float)t * __powf(10000.0f, -(float)d / 32.0f);
      outv = val * __cosf(th) - qkv[rowb + d + 32] * __sinf(th);
    } else {
      const int i = d - 32;
      const float th = (float)t * __powf(10000.0f, -(float)i / 32.0f);
      outv = qkv[rowb + i] * __sinf(th) + val * __cosf(th);
    }
    Kb[(size_t)hk * SEQ * HDIM + (size_t)t * HDIM + d] = (__bf16)outv;
  } else {                                   // ---- V, plain cast
    const int oo = o - (NH + NKV) * HDIM;
    const int hk = oo / HDIM, d = oo % HDIM;
    Vb[(size_t)hk * SEQ * HDIM + (size_t)t * HDIM + d] = (__bf16)val;
  }
}

// ---------------------------------------------------------------------------
// Causal flash attention, GQA (4 Q heads per KV head).
// Block = 128 threads (4 waves), 64 query rows (16 per wave); grid (S/64, NH).
// Each KV tile (32x64 K and V) is staged ONCE per block into LDS by the
// Tensor Data Mover (wave 0 issues tensor_load_to_lds, waits TENSORcnt),
// then all 4 waves consume it. Online softmax in fp32 accumulator layout.
// ---------------------------------------------------------------------------
__global__ void mha_flash(const __bf16* __restrict__ Q,
                          const __bf16* __restrict__ Kc,
                          const __bf16* __restrict__ V,
                          __bf16* __restrict__ ctx) {
  __shared__ __align__(16) __bf16 Ks[32][HDIM];     // KV-tile of K
  __shared__ __align__(16) __bf16 Vs[32][HDIM];     // KV-tile of V
  __shared__ __align__(16) __bf16 Plds[4][16][32];  // per-wave P staging

  const int wave = threadIdx.x >> 5;
  const int lane = threadIdx.x & 31;
  const int half = lane >> 4;
  const int ln   = lane & 15;

  const int h     = blockIdx.y;
  const int hkv   = h >> 2;                  // NH/NKV = 4
  const int brow0 = blockIdx.x * 64;         // block's first query row
  const int qrow0 = brow0 + wave * 16;       // wave's first query row

  const __bf16* Qh = Q + (size_t)h * SEQ * HDIM + (size_t)qrow0 * HDIM;
  const __bf16* Kh = Kc + (size_t)hkv * SEQ * HDIM;
  const __bf16* Vh = V + (size_t)hkv * SEQ * HDIM;

  const v16bf qa0 = load_frag_rowK(Qh, 0, HDIM, 0);
  const v16bf qa1 = load_frag_rowK(Qh, 0, HDIM, 32);

  v8f o0 = {}, o1 = {}, o2 = {}, o3 = {};
  float m[8], l[8];
#pragma unroll
  for (int r = 0; r < 8; ++r) { m[r] = -__builtin_inff(); l[r] = 0.0f; }

  const float scale = 0.125f;  // 1/sqrt(64)
  const unsigned ksOff = lds_offset_of(&Ks[0][0]);
  const unsigned vsOff = lds_offset_of(&Vs[0][0]);

  for (int kt = 0; kt < brow0 + 64; kt += 32) {
    __syncthreads();  // previous tile fully consumed
    if (wave == 0) {
#if HAVE_TDM
      tdm_load_tile_2d(ksOff, Kh + (size_t)kt * HDIM, 32, HDIM, HDIM);
      tdm_load_tile_2d(vsOff, Vh + (size_t)kt * HDIM, 32, HDIM, HDIM);
      tdm_wait0();
#else
      // cooperative fallback: 32 lanes copy 2x(32x64) bf16 tiles as uint4
      const uint4* gk = (const uint4*)(Kh + (size_t)kt * HDIM);
      const uint4* gv = (const uint4*)(Vh + (size_t)kt * HDIM);
      uint4* lk = (uint4*)&Ks[0][0];
      uint4* lv = (uint4*)&Vs[0][0];
#pragma unroll
      for (int e = 0; e < 8; ++e) {
        lk[lane + e * 32] = gk[lane + e * 32];
        lv[lane + e * 32] = gv[lane + e * 32];
      }
#endif
    }
    __syncthreads();  // tile visible to all waves

    // ---- scores: two 16x16 tiles over kv columns [kt, kt+32)
    v8f s0 = {}, s1 = {};
    s0 = wmma_bf16(qa0, load_frag_rowK(&Ks[0][0], 0,  HDIM, 0),  s0);
    s0 = wmma_bf16(qa1, load_frag_rowK(&Ks[0][0], 0,  HDIM, 32), s0);
    s1 = wmma_bf16(qa0, load_frag_rowK(&Ks[0][0], 16, HDIM, 0),  s1);
    s1 = wmma_bf16(qa1, load_frag_rowK(&Ks[0][0], 16, HDIM, 32), s1);

    // ---- online softmax (row = r + 8*half, col = lane&15)
#pragma unroll
    for (int r = 0; r < 8; ++r) {
      const int row = qrow0 + r + 8 * half;
      const int c0 = kt + ln;
      const int c1 = kt + 16 + ln;
      float v0 = (c0 <= row) ? s0[r] * scale : -__builtin_inff();
      float v1 = (c1 <= row) ? s1[r] * scale : -__builtin_inff();

      float mx = fmaxf(v0, v1);
      mx = fmaxf(mx, __shfl_xor(mx, 1, 32));
      mx = fmaxf(mx, __shfl_xor(mx, 2, 32));
      mx = fmaxf(mx, __shfl_xor(mx, 4, 32));
      mx = fmaxf(mx, __shfl_xor(mx, 8, 32));
      const float mn = fmaxf(m[r], mx);

      const float p0 = __expf(v0 - mn);
      const float p1 = __expf(v1 - mn);
      float sum = p0 + p1;
      sum += __shfl_xor(sum, 1, 32);
      sum += __shfl_xor(sum, 2, 32);
      sum += __shfl_xor(sum, 4, 32);
      sum += __shfl_xor(sum, 8, 32);

      const float f = __expf(m[r] - mn);
      l[r] = l[r] * f + sum;
      m[r] = mn;
      o0[r] *= f; o1[r] *= f; o2[r] *= f; o3[r] *= f;

      Plds[wave][r + 8 * half][ln]      = (__bf16)p0;
      Plds[wave][r + 8 * half][16 + ln] = (__bf16)p1;
    }

    // ---- PV: P (16x32) as A, V tiles (32x16 per 16 d-cols) as B (from LDS)
    const v16bf pa = load_frag_rowK(&Plds[wave][0][0], 0, 32, 0);
    o0 = wmma_bf16(pa, load_frag_colgather(&Vs[0][0], 0, HDIM, 0),  o0);
    o1 = wmma_bf16(pa, load_frag_colgather(&Vs[0][0], 0, HDIM, 16), o1);
    o2 = wmma_bf16(pa, load_frag_colgather(&Vs[0][0], 0, HDIM, 32), o2);
    o3 = wmma_bf16(pa, load_frag_colgather(&Vs[0][0], 0, HDIM, 48), o3);
  }

#pragma unroll
  for (int r = 0; r < 8; ++r) {
    const int row = qrow0 + r + 8 * half;
    const float inv = 1.0f / l[r];
    const size_t base = (size_t)row * HD + (size_t)h * HDIM + ln;
    ctx[base]      = (__bf16)(o0[r] * inv);
    ctx[base + 16] = (__bf16)(o1[r] * inv);
    ctx[base + 32] = (__bf16)(o2[r] * inv);
    ctx[base + 48] = (__bf16)(o3[r] * inv);
  }
}

// ---------------------------------------------------------------------------
extern "C" void kernel_launch(void* const* d_in, const int* in_sizes, int n_in,
                              void* d_out, int out_size, void* d_ws, size_t ws_size,
                              hipStream_t stream) {
  const float* hidden = (const float*)d_in[0];   // [1,S,E]
  const float* wqkv   = (const float*)d_in[1];   // [OPROJ,E]
  const float* bqkv   = (const float*)d_in[2];   // [OPROJ]
  const float* wout   = (const float*)d_in[3];   // [E,HD]
  const float* bout   = (const float*)d_in[4];   // [E]
  float* out = (float*)d_out;                    // [1,S,E] fp32

  char* ws = (char*)d_ws;
  size_t off = 0;
  __bf16* hiddenBF = (__bf16*)(ws + off); off += (size_t)SEQ * EMB * 2;
  __bf16* wqkvBF   = (__bf16*)(ws + off); off += (size_t)OPROJ * EMB * 2;
  __bf16* woutBF   = (__bf16*)(ws + off); off += (size_t)EMB * HD * 2;
  float*  qkv      = (float*)(ws + off);  off += (size_t)SEQ * OPROJ * 4;
  __bf16* Qb       = (__bf16*)(ws + off); off += (size_t)NH * SEQ * HDIM * 2;
  __bf16* Kb       = (__bf16*)(ws + off); off += (size_t)NKV * SEQ * HDIM * 2;
  __bf16* Vb       = (__bf16*)(ws + off); off += (size_t)NKV * SEQ * HDIM * 2;
  __bf16* ctx      = (__bf16*)(ws + off); off += (size_t)SEQ * HD * 2;
  (void)ws_size; (void)in_sizes; (void)n_in; (void)out_size;

  // 1) casts to bf16
  {
    int n = SEQ * EMB;
    mha_cvt_bf16<<<(n + 255) / 256, 256, 0, stream>>>(hidden, hiddenBF, n);
    n = OPROJ * EMB;
    mha_cvt_bf16<<<(n + 255) / 256, 256, 0, stream>>>(wqkv, wqkvBF, n);
    n = EMB * HD;
    mha_cvt_bf16<<<(n + 255) / 256, 256, 0, stream>>>(wout, woutBF, n);
  }

  // 2) QKV projection: [S,E] x [OPROJ,E]^T + bias -> fp32 [S,OPROJ]
  mha_gemm_bf16<<<dim3(OPROJ / 128, SEQ / 64), 128, 0, stream>>>(
      hiddenBF, wqkvBF, bqkv, qkv, SEQ, OPROJ, EMB);

  // 3) RoPE + split into head-major bf16 Q/K/V
  {
    const int n = SEQ * OPROJ;
    mha_rope_split<<<(n + 255) / 256, 256, 0, stream>>>(qkv, Qb, Kb, Vb);
  }

  // 4) causal flash attention (TDM-staged KV tiles) -> ctx bf16 [S, HD]
  mha_flash<<<dim3(SEQ / 64, NH), 128, 0, stream>>>(Qb, Kb, Vb, ctx);

  // 5) output projection: [S,HD] x [E,HD]^T + bias -> fp32 d_out
  mha_gemm_bf16<<<dim3(EMB / 128, SEQ / 64), 128, 0, stream>>>(
      ctx, woutBF, bout, out, SEQ, EMB, HD);
}
